// ClientModel_72559177499298
// MI455X (gfx1250) — matrix operand order repaired and checked by reading, not compiled, not run on hardware
//
#include <hip/hip_runtime.h>
#include <hip/hip_bf16.h>
#include <math.h>

// ---------------- model dims ----------------
#define BSZ   2048
#define TLEN  80
#define VOCAB 80
#define EDIM  8
#define HDIM  256
#define K1    288   // [x(8) | pad(24) | h1(256)]  -- 9 * 32
#define K2    512   // [h1(256) | h2(256)]         -- 16 * 32

typedef __attribute__((ext_vector_type(16))) __bf16 v16bf;
typedef __attribute__((ext_vector_type(8)))  float  v8f;
typedef __attribute__((ext_vector_type(4)))  int    v4i;

// ---- async global->LDS path (gfx1250), guarded so the file always compiles ----
#if __has_builtin(__builtin_amdgcn_global_load_async_to_lds_b128) && \
    __has_builtin(__builtin_amdgcn_s_wait_asynccnt)
#define USE_ASYNC_LDS 1
#else
#define USE_ASYNC_LDS 0
#endif

static __device__ __forceinline__ unsigned short f2bf(float f) {
  unsigned int u = __float_as_uint(f);
  u += 0x7FFFu + ((u >> 16) & 1u);   // round-to-nearest-even
  return (unsigned short)(u >> 16);
}
static __device__ __forceinline__ float sigm(float x) {
  return 1.0f / (1.0f + __expf(-x));
}
// tanh(x) = 1 - 2/(exp(2x)+1): exact limits at +/-inf, single v_exp_f32
static __device__ __forceinline__ float ftanh(float x) {
  float e = __expf(2.0f * x);
  return 1.0f - 2.0f / (e + 1.0f);
}

// ---------------- utility kernels ----------------
__global__ void zero_kernel(unsigned int* p, long n) {
  long i = (long)blockIdx.x * blockDim.x + threadIdx.x;
  long stride = (long)gridDim.x * blockDim.x;
  for (; i < n; i += stride) p[i] = 0u;
}

// Build transposed bf16 weights: Wt1 [1024 x 288], Wt2 [1024 x 512]
__global__ void prep_weights(const float* __restrict__ W1, const float* __restrict__ W2,
                             unsigned short* __restrict__ Wt1, unsigned short* __restrict__ Wt2) {
  int i = blockIdx.x * blockDim.x + threadIdx.x;
  const int n1 = 1024 * K1;
  if (i < n1) {
    int n = i / K1, k = i - n * K1;
    float v = 0.0f;
    if (k < EDIM)       v = W1[k * 1024 + n];          // x part
    else if (k >= 32)   v = W1[(k - 24) * 1024 + n];   // h part (rows 8..263)
    Wt1[n * K1 + k] = f2bf(v);
  } else {
    int j = i - n1;
    if (j < 1024 * K2) {
      int n = j / K2, k = j - n * K2;
      Wt2[n * K2 + k] = f2bf(W2[k * 1024 + n]);
    }
  }
}

// Per-timestep embedding gather into A1 cols 0..7
__global__ void gather_x(const int* __restrict__ feat, const float* __restrict__ emb,
                         unsigned short* __restrict__ A1p, int t) {
  int i = blockIdx.x * blockDim.x + threadIdx.x;   // 2048*8 threads
  int b = i >> 3, e = i & 7;
  int idx = feat[b * TLEN + t];
  A1p[(size_t)b * K1 + e] = f2bf(emb[idx * EDIM + e]);
}

// ---------------- fused LSTM step (WMMA bf16) ----------------
// Block: 256 threads (8 waves). Tile: 32 batch rows x 128 hidden cols.
// Wave w owns hidden cols [n0 + 16w, +16) for ALL 4 gates and 2 M-subtiles:
//   - each B fragment feeds 2 WMMAs (M reuse), each A fragment feeds 4 (gate reuse)
//   - i/j/f/o for one (row,col) land in the SAME lane -> cell update fully in
//     registers, no z spill, no second barrier.
template <int KK>
__global__ __launch_bounds__(256)
void lstm_step(const unsigned short* __restrict__ A,    // [BSZ x KK] bf16 activations
               const unsigned short* __restrict__ Wt,   // [1024 x KK] bf16 transposed weights
               const float* __restrict__ bias,          // [1024]
               float* __restrict__ C,                   // [BSZ x 256] cell state (in/out)
               unsigned short* __restrict__ dst1, int s1, int o1,  // h out (bf16)
               unsigned short* __restrict__ dst2, int s2, int o2,  // h out (bf16, optional)
               float* __restrict__ dstF)                // h out (fp32, optional)
{
  __shared__ unsigned short As[32 * KK];

  const int tid     = threadIdx.x;
  const int lane    = tid & 31;
  const int wv      = tid >> 5;
  const int rowBase = blockIdx.x * 32;                    // 64 row tiles
  const int col     = blockIdx.y * 128 + wv * 16 + (lane & 15); // this lane's hidden col
  const int hiHalf  = lane >> 4;                          // 0: lanes 0-15, 1: lanes 16-31

  // ---- stage A tile [32 x KK] to LDS; tile is one contiguous region ----
  const int nChunks = (32 * KK) >> 3;                     // 16B chunks
  const unsigned short* srcBase = A + (size_t)rowBase * KK;
#if USE_ASYNC_LDS
  {
    v4i* g = (v4i*)srcBase;        // builtin expects v4i* (16B granules)
    v4i* l = (v4i*)(&As[0]);
    for (int i = tid; i < nChunks; i += 256)
      __builtin_amdgcn_global_load_async_to_lds_b128(g + i, l + i, 0, 0);
    __builtin_amdgcn_s_wait_asynccnt(0);
  }
#else
  for (int i = tid; i < nChunks; i += 256)
    *reinterpret_cast<uint4*>(&As[i * 8]) =
        *reinterpret_cast<const uint4*>(srcBase + i * 8);
#endif
  __syncthreads();

  // ---- WMMA accumulation: acc[m][gate] ----
  v8f acc[2][4] = {};
  const unsigned short* wbase = Wt + (size_t)col * KK;    // gate g at + g*256*KK
#pragma unroll
  for (int k0 = 0; k0 < KK; k0 += 32) {
    const int kb = k0 + hiHalf * 16;
    v16bf b0 = *reinterpret_cast<const v16bf*>(wbase + (size_t)0 * 256 * KK + kb);
    v16bf b1 = *reinterpret_cast<const v16bf*>(wbase + (size_t)1 * 256 * KK + kb);
    v16bf b2 = *reinterpret_cast<const v16bf*>(wbase + (size_t)2 * 256 * KK + kb);
    v16bf b3 = *reinterpret_cast<const v16bf*>(wbase + (size_t)3 * 256 * KK + kb);
#pragma unroll
    for (int m = 0; m < 2; ++m) {
      const int ar = (m * 16 + (lane & 15)) * KK + k0 + hiHalf * 16;
      union { v16bf v; uint4 q[2]; } a;
      a.q[0] = *reinterpret_cast<const uint4*>(&As[ar]);
      a.q[1] = *reinterpret_cast<const uint4*>(&As[ar + 8]);
      acc[m][0] = __builtin_amdgcn_wmma_f32_16x16x32_bf16(false, a.v, false, b0, (short)0, acc[m][0], false, false);
      acc[m][1] = __builtin_amdgcn_wmma_f32_16x16x32_bf16(false, a.v, false, b1, (short)0, acc[m][1], false, false);
      acc[m][2] = __builtin_amdgcn_wmma_f32_16x16x32_bf16(false, a.v, false, b2, (short)0, acc[m][2], false, false);
      acc[m][3] = __builtin_amdgcn_wmma_f32_16x16x32_bf16(false, a.v, false, b3, (short)0, acc[m][3], false, false);
    }
  }

  // ---- fused cell update, entirely in registers (TF gate order i,j,f,o) ----
  const float bi  = bias[        col];
  const float bj  = bias[256   + col];
  const float bfg = bias[512   + col] + 1.0f;   // forget bias
  const float bo  = bias[768   + col];
#pragma unroll
  for (int m = 0; m < 2; ++m) {
#pragma unroll
    for (int v = 0; v < 8; ++v) {
      int row = rowBase + m * 16 + hiHalf * 8 + v;
      size_t ci = (size_t)row * HDIM + col;
      float iv = acc[m][0][v] + bi;
      float jv = acc[m][1][v] + bj;
      float fv = acc[m][2][v] + bfg;
      float ov = acc[m][3][v] + bo;
      float c  = C[ci];
      float cn = c * sigm(fv) + sigm(iv) * ftanh(jv);
      float hn = ftanh(cn) * sigm(ov);
      C[ci] = cn;
      unsigned short hb = f2bf(hn);
      dst1[(size_t)row * s1 + o1 + col] = hb;
      if (dst2) dst2[(size_t)row * s2 + o2 + col] = hb;
      if (dstF) dstF[(size_t)row * HDIM + col] = hn;
    }
  }
}

// ---------------- dense head: logits = h2 @ Wd + bd ----------------
__global__ void dense_kernel(const float* __restrict__ h, const float* __restrict__ Wd,
                             const float* __restrict__ bd, float* __restrict__ out) {
  int i = blockIdx.x * blockDim.x + threadIdx.x;
  if (i >= BSZ * VOCAB) return;
  int b = i / VOCAB, v = i - (i / VOCAB) * VOCAB;
  float acc = bd[v];
  const float* hp = h + (size_t)b * HDIM;
#pragma unroll 8
  for (int k = 0; k < HDIM; ++k) acc += hp[k] * Wd[k * VOCAB + v];
  out[i] = acc;
}

// ---------------- host orchestration ----------------
extern "C" void kernel_launch(void* const* d_in, const int* in_sizes, int n_in,
                              void* d_out, int out_size, void* d_ws, size_t ws_size,
                              hipStream_t stream) {
  const int*   feat = (const int*)d_in[0];
  const float* emb  = (const float*)d_in[1];
  const float* W1   = (const float*)d_in[2];
  const float* b1   = (const float*)d_in[3];
  const float* W2   = (const float*)d_in[4];
  const float* b2   = (const float*)d_in[5];
  const float* Wd   = (const float*)d_in[6];
  const float* bd   = (const float*)d_in[7];

  char* ws = (char*)d_ws;
  size_t off = 0;
  auto carve = [&](size_t bytes) -> void* {
    void* p = ws + off;
    off = (off + bytes + 255) & ~(size_t)255;
    return p;
  };
  unsigned short* Wt1   = (unsigned short*)carve((size_t)1024 * K1 * 2);
  unsigned short* Wt2   = (unsigned short*)carve((size_t)1024 * K2 * 2);
  unsigned short* A1[2] = { (unsigned short*)carve((size_t)BSZ * K1 * 2),
                            (unsigned short*)carve((size_t)BSZ * K1 * 2) };
  unsigned short* A2[2] = { (unsigned short*)carve((size_t)BSZ * K2 * 2),
                            (unsigned short*)carve((size_t)BSZ * K2 * 2) };
  float* c1  = (float*)carve((size_t)BSZ * HDIM * 4);
  float* c2  = (float*)carve((size_t)BSZ * HDIM * 4);
  float* h2f = (float*)carve((size_t)BSZ * HDIM * 4);

  // 1) bf16 transposed weights
  prep_weights<<<3200, 256, 0, stream>>>(W1, W2, Wt1, Wt2);

  // 2) zero A1[0..1], A2[0..1], c1, c2 (contiguous after Wt2; all 256B-multiple sizes)
  long zeroDwords = ((long)2 * BSZ * K1 * 2 + (long)2 * BSZ * K2 * 2 +
                     (long)2 * BSZ * HDIM * 4) / 4;
  zero_kernel<<<2048, 256, 0, stream>>>((unsigned int*)A1[0], zeroDwords);

  // 3) recurrence: ping-pong A buffers by timestep parity
  dim3 grid(BSZ / 32, HDIM / 128);
  for (int t = 0; t < TLEN; ++t) {
    int p = t & 1, q = p ^ 1;
    gather_x<<<(BSZ * EDIM) / 256, 256, 0, stream>>>(feat, emb, A1[p], t);
    // layer 1: reads A1[p]; h1 -> A1[q] cols 32.. (next t) and A2[p] cols 0.. (this t)
    lstm_step<K1><<<grid, 256, 0, stream>>>(A1[p], Wt1, b1, c1,
                                            A1[q], K1, 32,
                                            A2[p], K2, 0,
                                            nullptr);
    // layer 2: reads A2[p]; h2 -> A2[q] cols 256.. (next t) and fp32 h2f (head input)
    lstm_step<K2><<<grid, 256, 0, stream>>>(A2[p], Wt2, b2, c2,
                                            A2[q], K2, HDIM,
                                            nullptr, 0, 0,
                                            h2f);
  }

  // 4) dense head on final h2
  dense_kernel<<<(BSZ * VOCAB) / 256, 256, 0, stream>>>(h2f, Wd, bd, (float*)d_out);
}
